// GATv2Model_77292231459222
// MI455X (gfx1250) — compile-verified, matrix operand order checked by reading, not run on hardware
//
#include <hip/hip_runtime.h>
#include <hip/hip_bf16.h>
#include <stdint.h>

// ---------------------------------------------------------------------------
// GATv2 GNN for MI455X (gfx1250).  Dense GEMMs run on v_wmma_f32_16x16x32_bf16
// (bf16 in / f32 accumulate) with 16x64 register blocking per wave;
// attention softmax/scatter runs in f32 with hardware float atomics.
// wave32 throughout.
// ---------------------------------------------------------------------------

typedef __attribute__((ext_vector_type(16))) __bf16 v16bf;
typedef __attribute__((ext_vector_type(8)))  float  v8f;

#define GAT_H 8
#define GAT_C 64

__device__ __forceinline__ unsigned short f2bf(float f) {
  unsigned int u = __float_as_uint(f);
  u += 0x7FFFu + ((u >> 16) & 1u);          // round-to-nearest-even
  return (unsigned short)(u >> 16);
}

__device__ __forceinline__ void atomicMaxF(float* addr, float val) {
  // monotonic-bits trick: signed max for >=0, unsigned min for <0
  if (val >= 0.f) atomicMax((int*)addr, __float_as_int(val));
  else            atomicMin((unsigned int*)addr, __float_as_uint(val));
}

// -------------------------------- utility ----------------------------------

__global__ void k_fill_f32(float* p, float v, int n) {
  int i = blockIdx.x * blockDim.x + threadIdx.x;
  if (i < n) p[i] = v;
}

// f32 weight [K, Ncol] (row-major) -> bf16 [Ncol, Kpad] (row-major, K zero-pad)
__global__ void k_transpose_w_bf16(const float* __restrict__ w, unsigned short* __restrict__ o,
                                   int K, int Kpad, int Ncol) {
  int i = blockIdx.x * blockDim.x + threadIdx.x;
  if (i >= Ncol * Kpad) return;
  int n = i / Kpad, k = i - n * Kpad;
  o[i] = (k < K) ? f2bf(w[(size_t)k * Ncol + n]) : (unsigned short)0;
}

// f32 [M,K] -> bf16 [Mpad,Kpad], zero padding
__global__ void k_convert_pad_bf16(const float* __restrict__ x, unsigned short* __restrict__ o,
                                   int M, int Mpad, int K, int Kpad) {
  int i = blockIdx.x * blockDim.x + threadIdx.x;
  if (i >= Mpad * Kpad) return;
  int m = i / Kpad, k = i - m * Kpad;
  o[i] = (m < M && k < K) ? f2bf(x[(size_t)m * K + k]) : (unsigned short)0;
}

// o = bf16(a+b), padded rows zero (K == Kpad)
__global__ void k_add_convert_pad(const float* __restrict__ a, const float* __restrict__ b,
                                  unsigned short* __restrict__ o, int M, int Mpad, int K) {
  int i = blockIdx.x * blockDim.x + threadIdx.x;
  if (i >= Mpad * K) return;
  int m = i / K;
  o[i] = (m < M) ? f2bf(a[i] + b[i]) : (unsigned short)0;
}

// comb = concat([atom_msg, agg], dim=1) -> bf16 [Mpad, 128]
__global__ void k_concat_convert(const float* __restrict__ a, const float* __restrict__ b,
                                 unsigned short* __restrict__ o, int M, int Mpad) {
  int i = blockIdx.x * blockDim.x + threadIdx.x;
  if (i >= Mpad * 128) return;
  int m = i >> 7, k = i & 127;
  if (m >= M) { o[i] = 0; return; }
  float v = (k < 64) ? a[(size_t)m * 64 + k] : b[(size_t)m * 64 + (k - 64)];
  o[i] = f2bf(v);
}

// ------------------------------ WMMA GEMM ----------------------------------
// C[Mpad,Np] (f32) = A[Mpad,K](bf16) * B^T  where B stored as [Np,K](bf16).
// One wave computes a 16x64 strip: one A fragment is reused by 4 B fragments /
// 4 accumulators per K-slab (4x fewer A loads than tile-per-wave).
// K-loop in steps of 32 via v_wmma_f32_16x16x32_bf16.
// Fragment layouts per CDNA5 ISA 7.12.2:
//   16-bit A 16x32: lane L holds row M=L&15; VGPR r holds K pair
//     k = (r&3)*2 + (r>>2)*16 + (L>=16 ? 8 : 0)
//   B^T ([Np,K] rows are output columns) mirrors A's layout with N=L&15.
//   f32 C/D: VGPR r -> M = r + (L>=16 ? 8 : 0), N = L&15.
// Np must be a multiple of 64 (true for all layers: 64/128/512).
__global__ void k_gemm_bf16_wmma(const unsigned short* __restrict__ A,
                                 const unsigned short* __restrict__ B,
                                 const float* __restrict__ bias,
                                 float* __restrict__ Cg,
                                 int Mpad, int K, int Np) {
  const int lane  = threadIdx.x & 31;
  const int wave  = threadIdx.x >> 5;
  const int tileM = (blockIdx.y * 4 + wave) << 4;   // 16 rows per wave
  const int tileN = blockIdx.x << 6;                // 64 cols per wave
  if (tileM >= Mpad) return;                        // wave-uniform

  const int l15   = lane & 15;
  const int khalf = (lane >> 4) << 3;               // +0 or +8
  const unsigned short* ap  = A + (size_t)(tileM + l15) * K;
  const unsigned short* bp0 = B + (size_t)(tileN + l15) * K;
  const size_t bstride = (size_t)16 * K;            // next 16-col subtile

  v8f acc0 = {0.f,0.f,0.f,0.f,0.f,0.f,0.f,0.f};
  v8f acc1 = acc0, acc2 = acc0, acc3 = acc0;

  for (int k0 = 0; k0 < K; k0 += 32) {
    if (k0 + 32 < K) {
      __builtin_prefetch(ap + k0 + 32, 0, 1);       // global_prefetch_b8
      __builtin_prefetch(bp0 + k0 + 32, 0, 1);
    }
    union Frag { v16bf v; unsigned int u[8]; } fa, fb0, fb1, fb2, fb3;
#pragma unroll
    for (int r = 0; r < 8; ++r) {
      const int kk = k0 + khalf + ((r & 3) << 1) + ((r >> 2) << 4);
      fa.u[r]  = *(const unsigned int*)(ap + kk);
      fb0.u[r] = *(const unsigned int*)(bp0 + kk);
      fb1.u[r] = *(const unsigned int*)(bp0 + bstride + kk);
      fb2.u[r] = *(const unsigned int*)(bp0 + 2 * bstride + kk);
      fb3.u[r] = *(const unsigned int*)(bp0 + 3 * bstride + kk);
    }
    acc0 = __builtin_amdgcn_wmma_f32_16x16x32_bf16(false, fa.v, false, fb0.v,
                                                   (short)0, acc0, false, false);
    acc1 = __builtin_amdgcn_wmma_f32_16x16x32_bf16(false, fa.v, false, fb1.v,
                                                   (short)0, acc1, false, false);
    acc2 = __builtin_amdgcn_wmma_f32_16x16x32_bf16(false, fa.v, false, fb2.v,
                                                   (short)0, acc2, false, false);
    acc3 = __builtin_amdgcn_wmma_f32_16x16x32_bf16(false, fa.v, false, fb3.v,
                                                   (short)0, acc3, false, false);
  }

  // store 4 subtiles (+ optional bias broadcast over columns)
  float* crow = Cg + (size_t)(tileM + khalf) * Np + tileN + l15;
#pragma unroll
  for (int t = 0; t < 4; ++t) {
    const v8f acc = (t == 0) ? acc0 : (t == 1) ? acc1 : (t == 2) ? acc2 : acc3;
    const int cn = tileN + t * 16 + l15;
    const float bv = bias ? bias[cn] : 0.f;
#pragma unroll
    for (int r = 0; r < 8; ++r) {
      crow[(size_t)r * Np + t * 16] = acc[r] + bv;
    }
  }
}

// ------------------------------- BatchNorm ---------------------------------

__global__ void k_colstats(const float* __restrict__ x, float* __restrict__ stats,
                           int M, int cols) {
  int col = blockIdx.x * blockDim.x + threadIdx.x;
  if (col >= cols) return;
  int r0 = blockIdx.y * 128;
  int r1 = min(r0 + 128, M);
  float s1 = 0.f, s2 = 0.f;
  for (int r = r0; r < r1; ++r) {
    float v = x[(size_t)r * cols + col];
    s1 += v; s2 += v * v;
  }
  atomicAdd(&stats[col], s1);
  atomicAdd(&stats[cols + col], s2);
}

__global__ void k_bn_finalize(const float* __restrict__ stats, const float* __restrict__ g,
                              const float* __restrict__ be, float* __restrict__ scale,
                              float* __restrict__ shift, int cols, float invM) {
  int c = blockIdx.x * blockDim.x + threadIdx.x;
  if (c >= cols) return;
  float m = stats[c] * invM;
  float v = fmaxf(stats[cols + c] * invM - m * m, 0.f);
  float sc = g[c] * rsqrtf(v + 1e-5f);
  scale[c] = sc;
  shift[c] = be[c] - m * sc;
}

__global__ void k_bn_relu(float* __restrict__ x, const float* __restrict__ scale,
                          const float* __restrict__ shift, int M, int cols) {
  int i = blockIdx.x * blockDim.x + threadIdx.x;
  if (i >= M * cols) return;
  int c = i % cols;
  x[i] = fmaxf(scale[c] * x[i] + shift[c], 0.f);
}

__global__ void k_bn_relu_bf16(const float* __restrict__ x, const float* __restrict__ scale,
                               const float* __restrict__ shift, unsigned short* __restrict__ o,
                               int M, int Mpad, int cols) {
  int i = blockIdx.x * blockDim.x + threadIdx.x;
  if (i >= Mpad * cols) return;
  int m = i / cols, c = i - m * cols;
  if (m >= M) { o[i] = 0; return; }
  o[i] = f2bf(fmaxf(scale[c] * x[i] + shift[c], 0.f));
}

// ----------------------------- edge scatter --------------------------------

__global__ void k_edge_scatter_mean(const float* __restrict__ ef, const long long* __restrict__ col,
                                    float* __restrict__ sum, float* __restrict__ cnt, int E) {
  int gw   = blockIdx.x * (blockDim.x >> 5) + (threadIdx.x >> 5);
  int lane = threadIdx.x & 31;
  if (gw >= E) return;
  int c = (int)col[gw];
  const float* s = ef + (size_t)gw * 64;
  float* d = sum + (size_t)c * 64;
  atomicAdd(&d[lane],      s[lane]);
  atomicAdd(&d[lane + 32], s[lane + 32]);
  if (lane == 0) atomicAdd(&cnt[c], 1.f);
}

__global__ void k_div_cnt(float* __restrict__ sum, const float* __restrict__ cnt, int M) {
  int i = blockIdx.x * blockDim.x + threadIdx.x;
  if (i >= M * 64) return;
  sum[i] /= fmaxf(cnt[i >> 6], 1.f);
}

// ------------------------------ GATv2 kernels ------------------------------
// Edge list includes implicit self-loops: e < E uses row/col arrays,
// e >= E is the self-loop (e - E, e - E).

__global__ void k_gat_score(const float* __restrict__ xl, const float* __restrict__ xr,
                            const long long* __restrict__ row, const long long* __restrict__ col,
                            const float* __restrict__ att, float* __restrict__ score,
                            int E, int N) {
  int gw   = blockIdx.x * (blockDim.x >> 5) + (threadIdx.x >> 5);
  int lane = threadIdx.x & 31;
  int total = (E + N) * GAT_H;
  if (gw >= total) return;
  int e = gw >> 3, h = gw & 7;
  int r = (e < E) ? (int)row[e] : (e - E);
  int c = (e < E) ? (int)col[e] : (e - E);
  const float* pl = xl + (size_t)r * (GAT_H * GAT_C) + h * GAT_C;
  const float* pr = xr + (size_t)c * (GAT_H * GAT_C) + h * GAT_C;
  const float* pa = att + h * GAT_C;
  float s = 0.f;
#pragma unroll
  for (int k = 0; k < 2; ++k) {
    int cc = lane + k * 32;
    float v = pl[cc] + pr[cc];
    v = (v > 0.f) ? v : 0.2f * v;               // leaky_relu(0.2)
    s += v * pa[cc];
  }
#pragma unroll
  for (int off = 16; off > 0; off >>= 1) s += __shfl_xor(s, off, 32);
  if (lane == 0) score[gw] = s;
}

__global__ void k_seg_max(const float* __restrict__ score, const long long* __restrict__ col,
                          float* __restrict__ mbuf, int E, int N) {
  int i = blockIdx.x * blockDim.x + threadIdx.x;
  int total = (E + N) * GAT_H;
  if (i >= total) return;
  int e = i >> 3, h = i & 7;
  int c = (e < E) ? (int)col[e] : (e - E);
  atomicMaxF(&mbuf[c * GAT_H + h], score[i]);
}

__global__ void k_seg_expsum(float* __restrict__ score, const long long* __restrict__ col,
                             const float* __restrict__ mbuf, float* __restrict__ sbuf,
                             int E, int N) {
  int i = blockIdx.x * blockDim.x + threadIdx.x;
  int total = (E + N) * GAT_H;
  if (i >= total) return;
  int e = i >> 3, h = i & 7;
  int c = (e < E) ? (int)col[e] : (e - E);
  float ex = __expf(score[i] - mbuf[c * GAT_H + h]);
  score[i] = ex;                                 // overwrite with exp
  atomicAdd(&sbuf[c * GAT_H + h], ex);
}

__global__ void k_gat_aggregate(const float* __restrict__ xl, const float* __restrict__ ex,
                                const float* __restrict__ sbuf,
                                const long long* __restrict__ row, const long long* __restrict__ col,
                                float* __restrict__ out, int E, int N) {
  int gw   = blockIdx.x * (blockDim.x >> 5) + (threadIdx.x >> 5);
  int lane = threadIdx.x & 31;
  int total = (E + N) * GAT_H;
  if (gw >= total) return;
  int e = gw >> 3, h = gw & 7;
  int r = (e < E) ? (int)row[e] : (e - E);
  int c = (e < E) ? (int)col[e] : (e - E);
  float alpha = ex[gw] / sbuf[c * GAT_H + h];
  const float* pl = xl + (size_t)r * (GAT_H * GAT_C) + h * GAT_C;
  float* po = out + (size_t)c * (GAT_H * GAT_C) + h * GAT_C;
  atomicAdd(&po[lane],      alpha * pl[lane]);
  atomicAdd(&po[lane + 32], alpha * pl[lane + 32]);
}

// ------------------------------- final head --------------------------------

__global__ void k_final_dot(const float* __restrict__ q, const float* __restrict__ w,
                            const float* __restrict__ b, float* __restrict__ out, int M) {
  int gw   = blockIdx.x * (blockDim.x >> 5) + (threadIdx.x >> 5);
  int lane = threadIdx.x & 31;
  if (gw >= M) return;
  const float* p = q + (size_t)gw * 64;
  float s = p[lane] * w[lane] + p[lane + 32] * w[lane + 32];
#pragma unroll
  for (int off = 16; off > 0; off >>= 1) s += __shfl_xor(s, off, 32);
  if (lane == 0) out[gw] = s + b[0];
}

// ===========================================================================

extern "C" void kernel_launch(void* const* d_in, const int* in_sizes, int n_in,
                              void* d_out, int out_size, void* d_ws, size_t ws_size,
                              hipStream_t stream) {
  (void)n_in; (void)out_size; (void)ws_size;
  // ---- inputs (setup_inputs dict order; params in insertion order) ----
  const float* x     = (const float*)d_in[0];
  const float* ea    = (const float*)d_in[1];
  const float* ap_w  = (const float*)d_in[2];   const float* ap_b  = (const float*)d_in[3];
  const float* ap_g  = (const float*)d_in[4];   const float* ap_be = (const float*)d_in[5];
  const float* ep_w  = (const float*)d_in[6];   const float* ep_b  = (const float*)d_in[7];
  const float* ep_g  = (const float*)d_in[8];   const float* ep_be = (const float*)d_in[9];
  const float* am_w  = (const float*)d_in[10];  const float* am_b  = (const float*)d_in[11];
  const float* am_g  = (const float*)d_in[12];  const float* am_be = (const float*)d_in[13];
  const float* g1_wl = (const float*)d_in[14];  const float* g1_bl = (const float*)d_in[15];
  const float* g1_wr = (const float*)d_in[16];  const float* g1_br = (const float*)d_in[17];
  const float* g1_at = (const float*)d_in[18];  /* g1_bias d_in[19] cancels in BN1 */
  const float* bn1_g = (const float*)d_in[20];  const float* bn1_be= (const float*)d_in[21];
  const float* g2_wl = (const float*)d_in[22];  const float* g2_bl = (const float*)d_in[23];
  const float* g2_wr = (const float*)d_in[24];  const float* g2_br = (const float*)d_in[25];
  const float* g2_at = (const float*)d_in[26];  /* g2_bias d_in[27] cancels in BN2 */
  const float* bn2_g = (const float*)d_in[28];  const float* bn2_be= (const float*)d_in[29];
  const float* p1_w  = (const float*)d_in[30];  const float* p1_b  = (const float*)d_in[31];
  const float* p1_g  = (const float*)d_in[32];  const float* p1_be = (const float*)d_in[33];
  const float* p2_w  = (const float*)d_in[34];  const float* p2_b  = (const float*)d_in[35];
  const float* p2_g  = (const float*)d_in[36];  const float* p2_be = (const float*)d_in[37];
  const float* p3_w  = (const float*)d_in[38];  const float* p3_b  = (const float*)d_in[39];
  const long long* eidx = (const long long*)d_in[40];

  const int N = in_sizes[0] / 128;
  const int E = in_sizes[1] / 16;
  const long long* rowi = eidx;
  const long long* coli = eidx + E;
  const int Npad = (N + 15) & ~15;
  const int Epad = (E + 15) & ~15;
  float* out = (float*)d_out;

  // ---- workspace bump allocator (~165 MB total) ----
  size_t off = 0;
  auto alloc = [&](size_t bytes) -> char* {
    char* p = (char*)d_ws + off;
    off += (bytes + 255) & ~(size_t)255;
    return p;
  };
  unsigned short* Wap = (unsigned short*)alloc((size_t)64  * 128 * 2);
  unsigned short* Wep = (unsigned short*)alloc((size_t)64  * 32  * 2);
  unsigned short* Wam = (unsigned short*)alloc((size_t)64  * 64  * 2);
  unsigned short* W1l = (unsigned short*)alloc((size_t)512 * 128 * 2);
  unsigned short* W1r = (unsigned short*)alloc((size_t)512 * 128 * 2);
  unsigned short* W2l = (unsigned short*)alloc((size_t)512 * 512 * 2);
  unsigned short* W2r = (unsigned short*)alloc((size_t)512 * 512 * 2);
  unsigned short* Wp1 = (unsigned short*)alloc((size_t)128 * 512 * 2);
  unsigned short* Wp2 = (unsigned short*)alloc((size_t)64  * 128 * 2);
  unsigned short* Xb    = (unsigned short*)alloc((size_t)Npad * 128 * 2);
  unsigned short* Eb    = (unsigned short*)alloc((size_t)Epad * 32  * 2);
  unsigned short* t64b  = (unsigned short*)alloc((size_t)Npad * 64  * 2);
  unsigned short* combb = (unsigned short*)alloc((size_t)Npad * 128 * 2);
  unsigned short* hb    = (unsigned short*)alloc((size_t)Npad * 512 * 2);
  unsigned short* h2b   = (unsigned short*)alloc((size_t)Npad * 512 * 2);
  unsigned short* p128b = (unsigned short*)alloc((size_t)Npad * 128 * 2);
  float* af32  = (float*)alloc((size_t)Npad * 64  * 4);
  float* ef32  = (float*)alloc((size_t)Epad * 64  * 4);
  float* aggb  = (float*)alloc((size_t)N    * 64  * 4);
  float* cntb  = (float*)alloc((size_t)N          * 4);
  float* am32  = (float*)alloc((size_t)Npad * 64  * 4);
  float* xlb   = (float*)alloc((size_t)Npad * 512 * 4);
  float* xrb   = (float*)alloc((size_t)Npad * 512 * 4);
  float* hout  = (float*)alloc((size_t)Npad * 512 * 4);
  float* p128  = (float*)alloc((size_t)Npad * 128 * 4);
  float* p64   = (float*)alloc((size_t)Npad * 64  * 4);
  float* scoreb= (float*)alloc((size_t)(E + N) * GAT_H * 4);
  float* mb    = (float*)alloc((size_t)N * GAT_H * 4);
  float* sb    = (float*)alloc((size_t)N * GAT_H * 4);
  float* stats = (float*)alloc(1024 * 4);
  float* scaleb= (float*)alloc(512 * 4);
  float* shiftb= (float*)alloc(512 * 4);

  auto cdiv = [](int a, int b) { return (a + b - 1) / b; };

  // 16x64 strip per wave, 4 waves (64 rows) per block; Np % 64 == 0 always.
  auto gemm = [&](const unsigned short* A, const unsigned short* B, const float* bias,
                  float* Cg, int Mpad, int K, int Np) {
    dim3 g(Np / 64, cdiv(Mpad / 16, 4));
    k_gemm_bf16_wmma<<<g, 128, 0, stream>>>(A, B, bias, Cg, Mpad, K, Np);
  };
  auto bnstats = [&](const float* xv, int M, int cols, const float* g, const float* be) {
    k_fill_f32<<<cdiv(2 * cols, 256), 256, 0, stream>>>(stats, 0.f, 2 * cols);
    dim3 gr(cdiv(cols, 256), cdiv(M, 128));
    k_colstats<<<gr, 256, 0, stream>>>(xv, stats, M, cols);
    k_bn_finalize<<<cdiv(cols, 256), 256, 0, stream>>>(stats, g, be, scaleb, shiftb, cols, 1.f / M);
  };

  // ---- weight prep (bf16, transposed, K-padded) ----
  k_transpose_w_bf16<<<cdiv(64 * 128, 256), 256, 0, stream>>>(ap_w, Wap, 128, 128, 64);
  k_transpose_w_bf16<<<cdiv(64 * 32, 256), 256, 0, stream>>>(ep_w, Wep, 16, 32, 64);
  k_transpose_w_bf16<<<cdiv(64 * 64, 256), 256, 0, stream>>>(am_w, Wam, 64, 64, 64);
  k_transpose_w_bf16<<<cdiv(512 * 128, 256), 256, 0, stream>>>(g1_wl, W1l, 128, 128, 512);
  k_transpose_w_bf16<<<cdiv(512 * 128, 256), 256, 0, stream>>>(g1_wr, W1r, 128, 128, 512);
  k_transpose_w_bf16<<<cdiv(512 * 512, 256), 256, 0, stream>>>(g2_wl, W2l, 512, 512, 512);
  k_transpose_w_bf16<<<cdiv(512 * 512, 256), 256, 0, stream>>>(g2_wr, W2r, 512, 512, 512);
  k_transpose_w_bf16<<<cdiv(128 * 512, 256), 256, 0, stream>>>(p1_w, Wp1, 512, 512, 128);
  k_transpose_w_bf16<<<cdiv(64 * 128, 256), 256, 0, stream>>>(p2_w, Wp2, 128, 128, 64);

  // ---- atom preprocess: relu(BN(x @ ap_w + b)) ----
  k_convert_pad_bf16<<<cdiv(Npad * 128, 256), 256, 0, stream>>>(x, Xb, N, Npad, 128, 128);
  gemm(Xb, Wap, ap_b, af32, Npad, 128, 64);
  bnstats(af32, N, 64, ap_g, ap_be);
  k_bn_relu<<<cdiv(N * 64, 256), 256, 0, stream>>>(af32, scaleb, shiftb, N, 64);

  // ---- edge preprocess ----
  k_convert_pad_bf16<<<cdiv(Epad * 32, 256), 256, 0, stream>>>(ea, Eb, E, Epad, 16, 32);
  gemm(Eb, Wep, ep_b, ef32, Epad, 32, 64);
  bnstats(ef32, E, 64, ep_g, ep_be);
  k_bn_relu<<<cdiv(E * 64, 256), 256, 0, stream>>>(ef32, scaleb, shiftb, E, 64);

  // ---- scatter-mean of edge features onto cols ----
  k_fill_f32<<<cdiv(N * 64, 256), 256, 0, stream>>>(aggb, 0.f, N * 64);
  k_fill_f32<<<cdiv(N, 256), 256, 0, stream>>>(cntb, 0.f, N);
  k_edge_scatter_mean<<<cdiv(E, 8), 256, 0, stream>>>(ef32, coli, aggb, cntb, E);
  k_div_cnt<<<cdiv(N * 64, 256), 256, 0, stream>>>(aggb, cntb, N);

  // ---- atom message + comb = [atom_msg, agg] ----
  k_add_convert_pad<<<cdiv(Npad * 64, 256), 256, 0, stream>>>(af32, aggb, t64b, N, Npad, 64);
  gemm(t64b, Wam, am_b, am32, Npad, 64, 64);
  bnstats(am32, N, 64, am_g, am_be);
  k_bn_relu<<<cdiv(N * 64, 256), 256, 0, stream>>>(am32, scaleb, shiftb, N, 64);
  k_concat_convert<<<cdiv(Npad * 128, 256), 256, 0, stream>>>(am32, aggb, combb, N, Npad);

  // ---- one GATv2 layer (+BN+ReLU, repacked to bf16 for the next GEMM) ----
  const int tot = (E + N) * GAT_H;
  auto gat_layer = [&](const unsigned short* inb, int Kin,
                       const unsigned short* Wl, const unsigned short* Wr,
                       const float* bl, const float* br, const float* att,
                       const float* bng, const float* bnbe, unsigned short* outb) {
    gemm(inb, Wl, bl, xlb, Npad, Kin, 512);     // source transform
    gemm(inb, Wr, br, xrb, Npad, Kin, 512);     // target transform
    k_gat_score<<<cdiv(tot, 8), 256, 0, stream>>>(xlb, xrb, rowi, coli, att, scoreb, E, N);
    k_fill_f32<<<cdiv(N * GAT_H, 256), 256, 0, stream>>>(mb, -INFINITY, N * GAT_H);
    k_fill_f32<<<cdiv(N * GAT_H, 256), 256, 0, stream>>>(sb, 0.f, N * GAT_H);
    k_seg_max<<<cdiv(tot, 256), 256, 0, stream>>>(scoreb, coli, mb, E, N);
    k_seg_expsum<<<cdiv(tot, 256), 256, 0, stream>>>(scoreb, coli, mb, sb, E, N);
    k_fill_f32<<<cdiv(Npad * 512, 256), 256, 0, stream>>>(hout, 0.f, Npad * 512);
    k_gat_aggregate<<<cdiv(tot, 8), 256, 0, stream>>>(xlb, scoreb, sb, rowi, coli, hout, E, N);
    // (GAT output bias omitted: per-column shift cancels exactly in BatchNorm)
    bnstats(hout, N, 512, bng, bnbe);
    k_bn_relu_bf16<<<cdiv(Npad * 512, 256), 256, 0, stream>>>(hout, scaleb, shiftb, outb, N, Npad, 512);
  };
  gat_layer(combb, 128, W1l, W1r, g1_bl, g1_br, g1_at, bn1_g, bn1_be, hb);
  gat_layer(hb,    512, W2l, W2r, g2_bl, g2_br, g2_at, bn2_g, bn2_be, h2b);

  // ---- postprocess MLP ----
  gemm(h2b, Wp1, p1_b, p128, Npad, 512, 128);
  bnstats(p128, N, 128, p1_g, p1_be);
  k_bn_relu_bf16<<<cdiv(Npad * 128, 256), 256, 0, stream>>>(p128, scaleb, shiftb, p128b, N, Npad, 128);
  gemm(p128b, Wp2, p2_b, p64, Npad, 128, 64);
  bnstats(p64, N, 64, p2_g, p2_be);
  k_bn_relu<<<cdiv(N * 64, 256), 256, 0, stream>>>(p64, scaleb, shiftb, N, 64);
  k_final_dot<<<cdiv(N, 8), 256, 0, stream>>>(p64, p3_w, p3_b, out, N);
}
